// RefineUIGraphModule_90718299226373
// MI455X (gfx1250) — matrix operand order repaired and checked by reading, not compiled, not run on hardware
//
#include <hip/hip_runtime.h>

// ---------------------------------------------------------------------------
// Fused RefineUIGraph for MI455X (gfx1250, wave32, WMMA bf16 16x16x32).
//
// Per layer (L=2):  att = u @ seq^T -> L1 row-normalize -> msg = att@seq / S
//                   u = relu(u + msg @ W^T + b)
// Never materializes att (512MB > L2). Block = 8 waves owns 32 user rows:
//   pass1: waves split S 8-ways for att row norms (deterministic LDS combine)
//   pass2: per 256-sval chunk, waves cooperatively build normalized att tile
//          in LDS; each wave owns one 16-dim msg output tile (no atomics).
// 256 blocks x 8 waves = 2048 waves; each 1KB B tile feeds 2 WMMAs (M=32).
// ---------------------------------------------------------------------------

typedef __attribute__((ext_vector_type(16))) __bf16 v16bf;
typedef __attribute__((ext_vector_type(8)))  float  v8f;

#define U_DIM 8192
#define S_DIM 16384
#define D_DIM 128
#define EPS_P1 1e-12f

#define ATT_STRIDE 264   // 256 svals + 8 pad halves: stride 132 dwords -> 4-bank
                         // step per row, conflict-free b128 lane pattern
#define MSG_STRIDE 132   // f32 view of the same region (same byte stride)

__device__ __forceinline__ v8f wmma_bf16(v16bf a, v16bf b, v8f c) {
  return __builtin_amdgcn_wmma_f32_16x16x32_bf16(
      /*neg_a=*/false, a, /*neg_b=*/false, b,
      /*c_mod=*/(short)0, c, /*reuse_a=*/false, /*reuse_b=*/false);
}

// seq f32 [S,D] -> bf16 natural [S,D] and bf16 transposed [D,S]
__global__ void cvt_seq_kernel(const float* __restrict__ seq,
                               unsigned short* __restrict__ seq_bf_raw,
                               unsigned short* __restrict__ seqT_bf_raw) {
  int idx = blockIdx.x * blockDim.x + threadIdx.x;
  if (idx >= S_DIM * D_DIM) return;
  int s = idx >> 7;          // D = 128
  int d = idx & 127;
  __bf16 v = (__bf16)seq[idx];
  ((__bf16*)seq_bf_raw)[idx] = v;
  ((__bf16*)seqT_bf_raw)[(size_t)d * S_DIM + s] = v;
}

// One refinement layer. Block = 256 threads = 8 waves; block owns 32 users.
__global__ void __launch_bounds__(256)
refine_layer_kernel(const float* __restrict__ u_in,
                    const unsigned short* __restrict__ seq_bf_raw,
                    const unsigned short* __restrict__ seqT_bf_raw,
                    const float* __restrict__ Wmat,   // [L,D,D]
                    const float* __restrict__ bvec,   // [L,D]
                    float* __restrict__ u_out,
                    int layer) {
  const __bf16* seq_bf  = (const __bf16*)seq_bf_raw;   // [S][128]
  const __bf16* seqT_bf = (const __bf16*)seqT_bf_raw;  // [128][S]

  // 32 x ATT_STRIDE bf16 == 32 x MSG_STRIDE f32 == 16896 B  (region reused)
  __shared__ __align__(16) unsigned char smem[32 * ATT_STRIDE * 2 + 8 * 32 * 4];
  __bf16* att_lds = (__bf16*)smem;                       // [32][ATT_STRIDE]
  float*  msg_sh  = (float*)smem;                        // [32][MSG_STRIDE]
  float*  lds_rs  = (float*)(smem + 32 * ATT_STRIDE * 2);// [8][32] row sums

  const int tid  = threadIdx.x;
  const int w    = tid >> 5;          // wave 0..7
  const int lane = tid & 31;
  const int nn   = lane & 15;
  const int half = lane >> 4;
  const int m0   = blockIdx.x * 32;   // first user row of this block

  // ---- Load u tile [32 x 128] as bf16 A operands: a_u[uh][c] = 16x32 ----
  // A layout: halves 0..7 -> K = 16*half + i ; halves 8..15 -> K = 16*half+8+i
  v16bf a_u[2][4];
#pragma unroll
  for (int uh = 0; uh < 2; ++uh) {
    const float* urow = u_in + (size_t)(m0 + 16 * uh + nn) * D_DIM;
#pragma unroll
    for (int c = 0; c < 4; ++c) {
      const float* p0 = urow + 32 * c + 16 * half;
      const float* p1 = urow + 32 * c + 16 * half + 8;
#pragma unroll
      for (int i = 0; i < 8; ++i) {
        a_u[uh][c][i]     = (__bf16)p0[i];
        a_u[uh][c][8 + i] = (__bf16)p1[i];
      }
    }
  }

  // ---- Pass 1: att row L1 norms; wave w covers svals [w*2048, w*2048+2048) ----
  float rowabs[2][8];
#pragma unroll
  for (int uh = 0; uh < 2; ++uh)
#pragma unroll
    for (int r = 0; r < 8; ++r) rowabs[uh][r] = 0.0f;

  const int sBeg = w * (S_DIM / 8);
  const int sEnd = sBeg + (S_DIM / 8);
  for (int n0 = sBeg; n0 < sEnd; n0 += 16) {
    v16bf bt[4];
    const __bf16* sb = seq_bf + (size_t)(n0 + nn) * D_DIM + 16 * half;
#pragma unroll
    for (int c = 0; c < 4; ++c) bt[c] = *(const v16bf*)(sb + 32 * c);
#pragma unroll
    for (int uh = 0; uh < 2; ++uh) {
      v8f ca = {};
#pragma unroll
      for (int c = 0; c < 4; ++c) ca = wmma_bf16(a_u[uh][c], bt[c], ca);
#pragma unroll
      for (int r = 0; r < 8; ++r) rowabs[uh][r] += fabsf(ca[r]);
    }
  }

  // 16-lane butterfly, then publish per-wave row sums (fixed-order combine).
#pragma unroll
  for (int uh = 0; uh < 2; ++uh)
#pragma unroll
    for (int r = 0; r < 8; ++r) {
      float v = rowabs[uh][r];
      v += __shfl_xor(v, 1, 16);
      v += __shfl_xor(v, 2, 16);
      v += __shfl_xor(v, 4, 16);
      v += __shfl_xor(v, 8, 16);
      if (nn == 0) lds_rs[w * 32 + uh * 16 + 8 * half + r] = v;
    }
  __syncthreads();

  float inv[2][8];
#pragma unroll
  for (int uh = 0; uh < 2; ++uh)
#pragma unroll
    for (int r = 0; r < 8; ++r) {
      const int row = uh * 16 + 8 * half + r;
      float s = 0.0f;
      for (int w2 = 0; w2 < 8; ++w2) s += lds_rs[w2 * 32 + row];  // fixed order
      inv[uh][r] = 1.0f / fmaxf(s, EPS_P1);
    }
  __syncthreads();

  // ---- Pass 2: msg = norm(att) @ seq, chunked 256 svals at a time.
  // Wave w owns output dims [16w, 16w+16); c_msg[uh] accumulates its tile.
  v8f c_msg[2];
  c_msg[0] = (v8f){};
  c_msg[1] = (v8f){};

  for (int s0 = 0; s0 < S_DIM; s0 += 256) {
    // (a) wave w computes + normalizes att columns [s0+32w, s0+32w+32)
#pragma unroll
    for (int sub = 0; sub < 2; ++sub) {
      const int n0 = s0 + w * 32 + 16 * sub;
      v16bf bt[4];
      const __bf16* sb = seq_bf + (size_t)(n0 + nn) * D_DIM + 16 * half;
#pragma unroll
      for (int c = 0; c < 4; ++c) bt[c] = *(const v16bf*)(sb + 32 * c);
#pragma unroll
      for (int uh = 0; uh < 2; ++uh) {
        v8f ca = {};
#pragma unroll
        for (int c = 0; c < 4; ++c) ca = wmma_bf16(a_u[uh][c], bt[c], ca);
#pragma unroll
        for (int r = 0; r < 8; ++r) {
          const int mm = uh * 16 + r + 8 * half;
          att_lds[mm * ATT_STRIDE + w * 32 + 16 * sub + nn] =
              (__bf16)(ca[r] * inv[uh][r]);
        }
      }
    }
    __syncthreads();

    // (b) accumulate msg tile: A = att chunk (from LDS), B = seqT slice
#pragma unroll
    for (int kc = 0; kc < 8; ++kc) {
      const v16bf b2 = *(const v16bf*)(seqT_bf + (size_t)(16 * w + nn) * S_DIM +
                                       s0 + 32 * kc + 16 * half);
#pragma unroll
      for (int uh = 0; uh < 2; ++uh) {
        v16bf a2;
        const __bf16* ap = att_lds + (size_t)(nn + 16 * uh) * ATT_STRIDE + 32 * kc;
#pragma unroll
        for (int i = 0; i < 8; ++i) {
          a2[i]     = ap[16 * half + i];
          a2[8 + i] = ap[16 * half + 8 + i];
        }
        c_msg[uh] = wmma_bf16(a2, b2, c_msg[uh]);
      }
    }
    __syncthreads();   // protect att_lds before next chunk overwrites it
  }

  // ---- Projection: u' = relu(u + (msg/S) @ W^T + b) ----
  const float invS = 1.0f / (float)S_DIM;
#pragma unroll
  for (int uh = 0; uh < 2; ++uh)
#pragma unroll
    for (int r = 0; r < 8; ++r) {
      const int mm = uh * 16 + r + 8 * half;
      msg_sh[mm * MSG_STRIDE + 16 * w + nn] = c_msg[uh][r] * invS;
    }
  __syncthreads();

  // Wave w handles user-half uh = w&1 and output tiles t2 = w>>1, 4+(w>>1)
  const int uh  = w & 1;
  v16bf a_msg[4];
  {
    const float* mp = msg_sh + (size_t)(nn + 16 * uh) * MSG_STRIDE;
#pragma unroll
    for (int c = 0; c < 4; ++c) {
#pragma unroll
      for (int i = 0; i < 8; ++i) {
        a_msg[c][i]     = (__bf16)mp[32 * c + 16 * half + i];
        a_msg[c][8 + i] = (__bf16)mp[32 * c + 16 * half + 8 + i];
      }
    }
  }

  const float* Wl = Wmat + (size_t)layer * D_DIM * D_DIM;
  const float* bl = bvec + (size_t)layer * D_DIM;
#pragma unroll
  for (int tt = 0; tt < 2; ++tt) {
    const int t2 = (w >> 1) + 4 * tt;   // 8 waves x 2 -> all (uh, t2) pairs
    v8f co = {};
#pragma unroll
    for (int c = 0; c < 4; ++c) {
      // B[k][n] = W[16*t2 + n][k]  (msg @ W^T): W row is the B column
      const float* wp = Wl + (size_t)(16 * t2 + nn) * D_DIM + 32 * c + 16 * half;
      v16bf bw;
#pragma unroll
      for (int i = 0; i < 16; ++i) bw[i] = (__bf16)wp[i];
      co = wmma_bf16(a_msg[c], bw, co);
    }
    const int d  = 16 * t2 + nn;
    const float bb = bl[d];
#pragma unroll
    for (int r = 0; r < 8; ++r) {
      const int mm = m0 + 16 * uh + r + 8 * half;
      float val = co[r] + u_in[(size_t)mm * D_DIM + d] + bb;
      u_out[(size_t)mm * D_DIM + d] = fmaxf(val, 0.0f);
    }
  }
}

extern "C" void kernel_launch(void* const* d_in, const int* in_sizes, int n_in,
                              void* d_out, int out_size, void* d_ws, size_t ws_size,
                              hipStream_t stream) {
  const float* user_feat = (const float*)d_in[0];  // [U,D]
  const float* seq_feat  = (const float*)d_in[1];  // [S,D]
  const float* Wmat      = (const float*)d_in[2];  // [L,D,D]
  const float* bvec      = (const float*)d_in[3];  // [L,D]
  float* out = (float*)d_out;                      // [U,D]

  // Workspace layout: seq_bf16 (4MB) | seqT_bf16 (4MB) | u_mid f32 (4MB)
  char* ws = (char*)d_ws;
  unsigned short* seq_bf  = (unsigned short*)(ws);
  unsigned short* seqT_bf = (unsigned short*)(ws + ((size_t)4 << 20));
  float*          u_mid   = (float*)        (ws + ((size_t)8 << 20));

  const int cvt_threads = 256;
  const int cvt_blocks  = (S_DIM * D_DIM + cvt_threads - 1) / cvt_threads;
  cvt_seq_kernel<<<cvt_blocks, cvt_threads, 0, stream>>>(seq_feat, seq_bf, seqT_bf);

  // 256 blocks x 8 waves x (32 users / block) = 8192 users, 2048 waves
  refine_layer_kernel<<<256, 256, 0, stream>>>(user_feat, seq_bf, seqT_bf,
                                               Wmat, bvec, u_mid, 0);
  refine_layer_kernel<<<256, 256, 0, stream>>>(u_mid, seq_bf, seqT_bf,
                                               Wmat, bvec, out, 1);
}
